// StatefulMaskedConvLayer_15788299780146
// MI455X (gfx1250) — compile-verified, matrix operand order checked by reading, not run on hardware
//
#include <hip/hip_runtime.h>

typedef __attribute__((ext_vector_type(16))) _Float16 v16h;
typedef __attribute__((ext_vector_type(8)))  _Float16 v8h;
typedef __attribute__((ext_vector_type(8)))  float    v8f;
typedef __attribute__((ext_vector_type(4)))  float    f32x4;

#define NSTATES  8
#define NFILT    64
#define CIN      64
#define BB       32
#define HH       64
#define WW       64
#define KTILES   24          // K = 12 taps * 64 ch = 768 = 24 * 32
#define NTILES   4           // N = 64 = 4 * 16
#define BPACK_ELEMS (NSTATES * NTILES * KTILES * 32 * 16)   // 393216 halves
#define BPACK_BYTES ((size_t)BPACK_ELEMS * 2)               // 786432 B (256-aligned)
#define X_ELEMS     ((size_t)BB * HH * WW * CIN)            // 8388608
#define XH_BYTES    (X_ELEMS * 2)                           // 16 MiB

// ---------------------------------------------------------------------------
// Prologue 1: pack masked filters (only the 12 causally-valid taps) into the
// exact per-lane VGPR fragment order for V_WMMA_F32_16X16X32_F16's B operand:
// layout [state][ntile][ktile][lane][e], 16 halves per lane per k-tile.
// ---------------------------------------------------------------------------
__global__ __launch_bounds__(256)
void pack_filters_kernel(const float* __restrict__ filt, _Float16* __restrict__ bp) {
    int idx = blockIdx.x * 256 + threadIdx.x;
    if (idx >= BPACK_ELEMS) return;
    int e    = idx & 15;
    int lane = (idx >> 4) & 31;
    int rem  = idx >> 9;
    int kt   = rem % KTILES;  rem /= KTILES;
    int nt   = rem % NTILES;
    int s    = rem / NTILES;

    // B 16-bit fragment layout (mirror of A): lanes 0-15 carry K={0..7,16..23},
    // lanes 16-31 carry K={8..15,24..31} of column N=lane%16.
    int klo = (lane < 16) ? 0 : 8;
    int kin = (e < 8) ? (klo + e) : (16 + klo + (e - 8));
    int k   = kt * 32 + kin;           // 0..767
    int t   = k >> 6;                  // tap 0..11 (causally valid taps only)
    int c   = k & 63;                  // input channel
    int fy  = (t < 5) ? 0 : ((t < 10) ? 1 : 2);
    int fx  = (t < 5) ? t : ((t < 10) ? (t - 5) : (t - 10));
    int f   = nt * 16 + (lane & 15);   // output filter 0..63
    int o   = s * NFILT + f;           // channel in [fs,fs,Cin,S*F]
    float v = filt[(((fy * 5 + fx) * CIN + c) * (NSTATES * NFILT)) + o];
    bp[idx] = (_Float16)v;
}

// ---------------------------------------------------------------------------
// Prologue 2: convert x to f16 once (halves hot-loop bytes, kills in-loop cvt).
// 8 elements per thread, vector loads/stores.
// ---------------------------------------------------------------------------
__global__ __launch_bounds__(256)
void cvt_x_kernel(const float* __restrict__ x, _Float16* __restrict__ xh) {
    size_t i = ((size_t)blockIdx.x * 256 + threadIdx.x) * 8;
    if (i >= X_ELEMS) return;
    f32x4 a = *(const f32x4*)(x + i);
    f32x4 b = *(const f32x4*)(x + i + 4);
    v8h r;
    r[0] = (_Float16)a.x; r[1] = (_Float16)a.y; r[2] = (_Float16)a.z; r[3] = (_Float16)a.w;
    r[4] = (_Float16)b.x; r[5] = (_Float16)b.y; r[6] = (_Float16)b.z; r[7] = (_Float16)b.w;
    *(v8h*)(xh + i) = r;
}

// ---------------------------------------------------------------------------
// Main kernel: one wave per 16-pixel tile (row-aligned), 4 waves per block.
// Grid: B*H*(W/16) = 8192 tiles -> 2048 blocks of 128 threads.
// M=16 pixels, N=64 filters (4 n-tiles), K=768 (24 k-tiles of 32).
// ---------------------------------------------------------------------------
template <bool USE_F16X>
__global__ __launch_bounds__(128)
void conv_wmma_kernel(const float* __restrict__ x,
                      const _Float16* __restrict__ xh,
                      const int*   __restrict__ st,
                      const _Float16* __restrict__ bp,
                      const float* __restrict__ bias,
                      float* __restrict__ out) {
    const int wave = threadIdx.x >> 5;
    const int lane = threadIdx.x & 31;
    const int tile = blockIdx.x * 4 + wave;          // 0..8191
    const int wt   = tile & 3;
    const int h    = (tile >> 2) & 63;
    const int b    = tile >> 8;
    const int w0   = wt * 16;

    const int s   = st[b];
    const int m   = lane & 15;                       // A-row this lane serves
    const int klo = (lane < 16) ? 0 : 8;

    v8f acc0 = {}, acc1 = {}, acc2 = {}, acc3 = {};

    const _Float16* bbase = bp + (size_t)s * (NTILES * KTILES * 32 * 16);
    const size_t ntstride = (size_t)KTILES * 512;    // halves per ntile block

#pragma unroll 2
    for (int kt = 0; kt < KTILES; ++kt) {
        const int t   = kt >> 1;                     // tap index 0..11
        const int ch0 = (kt & 1) * 32;               // channel half
        const int fy  = (t < 5) ? 0 : ((t < 10) ? 1 : 2);
        const int fx  = (t < 5) ? t : ((t < 10) ? (t - 5) : (t - 10));
        const int hs  = h + fy - 2;
        const int ws  = w0 + m + fx - 2;

        v16h a = {};
        const bool inb = (hs >= 0) & (hs < HH) & (ws >= 0) & (ws < WW);
        if (inb) {
            const size_t pix = (((size_t)b * HH + hs) * WW + ws) * CIN + ch0 + klo;
            if (USE_F16X) {
                v8h lo = *(const v8h*)(xh + pix);        // K run {klo..klo+7}
                v8h hi = *(const v8h*)(xh + pix + 16);   // K run {16+klo..}
                a = __builtin_shufflevector(lo, hi, 0, 1, 2, 3, 4, 5, 6, 7,
                                                    8, 9, 10, 11, 12, 13, 14, 15);
            } else {
                const float* xb = x + pix;
                f32x4 f0 = *(const f32x4*)(xb);
                f32x4 f1 = *(const f32x4*)(xb + 4);
                f32x4 f2 = *(const f32x4*)(xb + 16);
                f32x4 f3 = *(const f32x4*)(xb + 20);
                a[0]  = (_Float16)f0.x; a[1]  = (_Float16)f0.y;
                a[2]  = (_Float16)f0.z; a[3]  = (_Float16)f0.w;
                a[4]  = (_Float16)f1.x; a[5]  = (_Float16)f1.y;
                a[6]  = (_Float16)f1.z; a[7]  = (_Float16)f1.w;
                a[8]  = (_Float16)f2.x; a[9]  = (_Float16)f2.y;
                a[10] = (_Float16)f2.z; a[11] = (_Float16)f2.w;
                a[12] = (_Float16)f3.x; a[13] = (_Float16)f3.y;
                a[14] = (_Float16)f3.z; a[15] = (_Float16)f3.w;
            }
        }

        const size_t boff = (size_t)kt * 512 + (size_t)lane * 16;
        v16h b0v = *(const v16h*)(bbase + boff);
        v16h b1v = *(const v16h*)(bbase + ntstride * 1 + boff);
        v16h b2v = *(const v16h*)(bbase + ntstride * 2 + boff);
        v16h b3v = *(const v16h*)(bbase + ntstride * 3 + boff);

        acc0 = __builtin_amdgcn_wmma_f32_16x16x32_f16(false, a, false, b0v, (short)0, acc0, false, false);
        acc1 = __builtin_amdgcn_wmma_f32_16x16x32_f16(false, a, false, b1v, (short)0, acc1, false, false);
        acc2 = __builtin_amdgcn_wmma_f32_16x16x32_f16(false, a, false, b2v, (short)0, acc2, false, false);
        acc3 = __builtin_amdgcn_wmma_f32_16x16x32_f16(false, a, false, b3v, (short)0, acc3, false, false);
    }

    // Epilogue: bias + ReLU.  C/D layout: VGPR r -> M = r + 8*(lane>=16), N = lane&15.
    const int fl   = lane & 15;
    const int mrow = (lane >> 4) * 8;
    const float bv0 = bias[s * NFILT +  0 + fl];
    const float bv1 = bias[s * NFILT + 16 + fl];
    const float bv2 = bias[s * NFILT + 32 + fl];
    const float bv3 = bias[s * NFILT + 48 + fl];

#pragma unroll
    for (int r = 0; r < 8; ++r) {
        const int mm = mrow + r;
        const size_t o = (((size_t)b * HH + h) * WW + (w0 + mm)) * NFILT;
        float v0 = acc0[r] + bv0;  out[o +  0 + fl] = v0 > 0.f ? v0 : 0.f;
        float v1 = acc1[r] + bv1;  out[o + 16 + fl] = v1 > 0.f ? v1 : 0.f;
        float v2 = acc2[r] + bv2;  out[o + 32 + fl] = v2 > 0.f ? v2 : 0.f;
        float v3 = acc3[r] + bv3;  out[o + 48 + fl] = v3 > 0.f ? v3 : 0.f;
    }
}

extern "C" void kernel_launch(void* const* d_in, const int* in_sizes, int n_in,
                              void* d_out, int out_size, void* d_ws, size_t ws_size,
                              hipStream_t stream) {
    const float* x       = (const float*)d_in[0];   // [32,64,64,64] f32
    const int*   st      = (const int*)  d_in[1];   // [32] i32
    const float* filters = (const float*)d_in[2];   // [5,5,64,512] f32
    const float* bias    = (const float*)d_in[3];   // [1,512] f32

    _Float16* bp = (_Float16*)d_ws;                              // packed B fragments
    _Float16* xh = (_Float16*)((char*)d_ws + BPACK_BYTES);       // f16 copy of x

    pack_filters_kernel<<<(BPACK_ELEMS + 255) / 256, 256, 0, stream>>>(filters, bp);

    if (ws_size >= BPACK_BYTES + XH_BYTES) {
        cvt_x_kernel<<<(int)((X_ELEMS / 8 + 255) / 256), 256, 0, stream>>>(x, xh);
        conv_wmma_kernel<true><<<2048, 128, 0, stream>>>(x, xh, st, bp, bias, (float*)d_out);
    } else {
        conv_wmma_kernel<false><<<2048, 128, 0, stream>>>(x, xh, st, bp, bias, (float*)d_out);
    }
}
